// GGMLLayer_23493471109657
// MI455X (gfx1250) — matrix operand order repaired and checked by reading, not compiled, not run on hardware
//
#include <hip/hip_runtime.h>

// ---- problem constants (match reference) ----
#define M_TOT 4096     // B*S
#define N_TOT 16384    // D_OUT
#define K_TOT 4096     // D_IN
#define NBLKS 128      // K_TOT / 32
#define NKB   128      // K iterations (K_TOT / BK)

// ---- tiling ----
#define BM 128
#define BN 128
#define BK 32
#define LDSS 40        // bf16 elements per LDS row (80B: 16B aligned, conflict-free)

typedef __attribute__((ext_vector_type(4)))  float   f32x4;
typedef __attribute__((ext_vector_type(4)))  int     i32x4;
typedef __attribute__((ext_vector_type(4)))  __bf16  bf16x4;
typedef __attribute__((ext_vector_type(8)))  __bf16  bf16x8;
typedef __attribute__((ext_vector_type(16))) __bf16  bf16x16;
typedef __attribute__((ext_vector_type(8)))  float   f32x8;

__device__ __forceinline__ bf16x16 load_frag(const __bf16* p0, const __bf16* p1) {
    bf16x8 lo = *reinterpret_cast<const bf16x8*>(p0);   // ds_load_b128
    bf16x8 hi = *reinterpret_cast<const bf16x8*>(p1);   // ds_load_b128
    bf16x16 r;
#pragma unroll
    for (int j = 0; j < 8; ++j) { r[j] = lo[j]; r[8 + j] = hi[j]; }
    return r;
}

__global__ __launch_bounds__(256, 2) void q8_gemm_bf16_wmma(
    const float* __restrict__ x,       // [M_TOT, K_TOT] fp32
    const int*   __restrict__ qs,      // [N_TOT, NBLKS, 32] int32 (int8 values)
    const float* __restrict__ scales,  // [N_TOT, NBLKS]
    const float* __restrict__ bias,    // [N_TOT]
    float*       __restrict__ out)     // [M_TOT, N_TOT]
{
    __shared__ __align__(16) __bf16 ldsA[2][BM * LDSS];
    __shared__ __align__(16) __bf16 ldsB[2][BN * LDSS];

    const int tid   = threadIdx.x;
    const int lane  = tid & 31;
    const int wave  = tid >> 5;
    const int waveM = wave & 3;   // 4 waves along M (32 rows each)
    const int waveN = wave >> 2;  // 2 waves along N (64 cols each)
    const int mBase = blockIdx.y * BM;
    const int nBase = blockIdx.x * BN;

    f32x8 acc[2][4];
#pragma unroll
    for (int mi = 0; mi < 2; ++mi)
#pragma unroll
        for (int ni = 0; ni < 4; ++ni)
#pragma unroll
            for (int j = 0; j < 8; ++j) acc[mi][ni][j] = 0.0f;

    // --- staging registers for the next K-slice (double buffering) ---
    f32x4 pa[4];
    i32x4 pb[4];
    float ps[4];

    // each thread moves 4 x float4 of A and 4 x int4 of B per K-slice:
    //   flat chunk f = tid + 256*i ; row = f/8 ; k4 = f%8 (4-elem groups of BK=32)
    auto load_global = [&](int kb) {
#pragma unroll
        for (int i = 0; i < 4; ++i) {
            const int f   = tid + (i << 8);
            const int row = f >> 3;
            const int k4  = (f & 7) << 2;
            pa[i] = *reinterpret_cast<const f32x4*>(
                x + (size_t)(mBase + row) * K_TOT + (size_t)kb * BK + k4);
            const int o = nBase + row;
            pb[i] = *reinterpret_cast<const i32x4*>(
                qs + (size_t)o * K_TOT + (size_t)kb * BK + k4);
            ps[i] = scales[(size_t)o * NBLKS + kb];
        }
    };

    auto store_lds = [&](int buf) {
#pragma unroll
        for (int i = 0; i < 4; ++i) {
            const int f   = tid + (i << 8);
            const int row = f >> 3;
            const int k4  = (f & 7) << 2;
            bf16x4 va, vb;
#pragma unroll
            for (int j = 0; j < 4; ++j) {
                va[j] = (__bf16)pa[i][j];
                vb[j] = (__bf16)(ps[i] * (float)pb[i][j]);  // dequantize Q8_0
            }
            *reinterpret_cast<bf16x4*>(&ldsA[buf][row * LDSS + k4]) = va; // ds_store_b64
            *reinterpret_cast<bf16x4*>(&ldsB[buf][row * LDSS + k4]) = vb; // ds_store_b64
        }
    };

    load_global(0);
    store_lds(0);
    __syncthreads();

    const int half = (lane < 16) ? 0 : 1;
    const int rsub = lane & 15;

    for (int kb = 0; kb < NKB; ++kb) {
        const int cur = kb & 1;
        if (kb + 1 < NKB) load_global(kb + 1);

        // ---- fragment loads per CDNA5 16-bit WMMA VGPR layouts ----
        bf16x16 afr[2], bfr[4];
        const int kA = half ? 8 : 0;    // A: lanes16-31 -> K 8..15 / 24..31
        const int kB = half ? 8 : 0;    // B: lanes16-31 -> K 8..15 / 24..31
#pragma unroll
        for (int mi = 0; mi < 2; ++mi) {
            const __bf16* rA = &ldsA[cur][(waveM * 32 + mi * 16 + rsub) * LDSS];
            afr[mi] = load_frag(rA + kA, rA + kA + 16);
        }
#pragma unroll
        for (int ni = 0; ni < 4; ++ni) {
            const __bf16* rB = &ldsB[cur][(waveN * 64 + ni * 16 + rsub) * LDSS];
            bfr[ni] = load_frag(rB + kB, rB + kB + 16);
        }

#pragma unroll
        for (int mi = 0; mi < 2; ++mi)
#pragma unroll
            for (int ni = 0; ni < 4; ++ni)
                acc[mi][ni] = __builtin_amdgcn_wmma_f32_16x16x32_bf16(
                    false, afr[mi], false, bfr[ni],
                    (short)0, acc[mi][ni], false, false);

        if (kb + 1 < NKB) store_lds(cur ^ 1);
        __syncthreads();
    }

    // ---- epilogue: bias add + store (C/D layout: VGPR i -> row half*8+i, col lane&15)
#pragma unroll
    for (int ni = 0; ni < 4; ++ni) {
        const int c  = nBase + waveN * 64 + ni * 16 + rsub;
        const float bv = bias[c];
#pragma unroll
        for (int mi = 0; mi < 2; ++mi) {
            const int r0 = mBase + waveM * 32 + mi * 16 + half * 8;
#pragma unroll
            for (int i = 0; i < 8; ++i)
                out[(size_t)(r0 + i) * N_TOT + c] = acc[mi][ni][i] + bv;
        }
    }
}

extern "C" void kernel_launch(void* const* d_in, const int* in_sizes, int n_in,
                              void* d_out, int out_size, void* d_ws, size_t ws_size,
                              hipStream_t stream) {
    (void)in_sizes; (void)n_in; (void)out_size; (void)d_ws; (void)ws_size;
    const float* x      = (const float*)d_in[0];
    const int*   qs     = (const int*)d_in[1];
    const float* scales = (const float*)d_in[2];
    const float* bias   = (const float*)d_in[3];
    float*       out    = (float*)d_out;

    dim3 grid(N_TOT / BN, M_TOT / BM);  // 128 x 32 workgroups
    q8_gemm_bf16_wmma<<<grid, 256, 0, stream>>>(x, qs, scales, bias, out);
}
